// ARITMPoolModel_67104569033189
// MI455X (gfx1250) — matrix operand order repaired and checked by reading, not compiled, not run on hardware
//
#include <hip/hip_runtime.h>

// ---------------------------------------------------------------------------
// Types for CDNA5 WMMA
// ---------------------------------------------------------------------------
typedef __attribute__((ext_vector_type(16))) __bf16 v16bf;
typedef __attribute__((ext_vector_type(8)))  float  v8f;

#define Bsz   32
#define Nn    1024
#define DIN   768
#define Hh    1024
#define KSEL  512

// ---------------------------------------------------------------------------
// bf16 helpers (round-to-nearest-even)
// ---------------------------------------------------------------------------
__device__ __forceinline__ unsigned short f2bf(float f) {
    union { float f; unsigned u; } x; x.f = f;
    unsigned r = x.u + 0x7FFFu + ((x.u >> 16) & 1u);
    return (unsigned short)(r >> 16);
}
__device__ __forceinline__ float bf2f(unsigned short h) {
    union { unsigned u; float f; } x; x.u = ((unsigned)h) << 16;
    return x.f;
}

// ---------------------------------------------------------------------------
// f32 -> bf16 elementwise conversion (activations, row-major kept)
// ---------------------------------------------------------------------------
__global__ void cvt_bf16_kernel(const float* __restrict__ in,
                                unsigned short* __restrict__ out, long long n) {
    long long i = (long long)blockIdx.x * blockDim.x + threadIdx.x;
    long long stride = (long long)gridDim.x * blockDim.x;
    for (; i < n; i += stride) out[i] = f2bf(in[i]);
}

// f32 W[K,N] -> bf16 W^T[N,K]  (so GEMM B-tiles are contiguous in K)
__global__ __launch_bounds__(256)
void transpose_cvt_kernel(const float* __restrict__ W, unsigned short* __restrict__ Wt,
                          int K, int N) {
    __shared__ float tile[32][33];
    int nb = blockIdx.x * 32, kb = blockIdx.y * 32;
    int tx = threadIdx.x & 31, ty = threadIdx.x >> 5;   // 32 x 8
    #pragma unroll
    for (int i = 0; i < 32; i += 8)
        tile[ty + i][tx] = W[(long long)(kb + ty + i) * N + nb + tx];
    __syncthreads();
    #pragma unroll
    for (int i = 0; i < 32; i += 8)
        Wt[(long long)(nb + ty + i) * K + kb + tx] = f2bf(tile[tx][ty + i]);
}

__global__ void zero_kernel(float* p, int n) {
    int i = blockIdx.x * 256 + threadIdx.x;
    if (i < n) p[i] = 0.0f;
}

// ---------------------------------------------------------------------------
// WMMA bf16 GEMM:  C[M,1024] = A[M,K] * B[K,1024] + bias,  B given as B^T[1024,K]
//   EPI 0: store f32 (pre-BN raw output)
//   EPI 1: leaky_relu(0.2), store bf16 (attention hidden hh)
//   GATHER: A row r -> x1b[b*1024 + idx[b*512 + i]] with b=r>>9, i=r&511
// Block: 256 thr (8 waves), tile 128x128, BK=32, double-buffered LDS.
// Wave = 32x64 = 2x4 WMMA accumulators.
// ---------------------------------------------------------------------------
template<int EPI, bool GATHER>
__global__ __launch_bounds__(256, 2)
void gemm_bf16_kernel(const unsigned short* __restrict__ A,
                      const unsigned short* __restrict__ Bt,
                      const float* __restrict__ bias,
                      float* __restrict__ Cf,
                      unsigned short* __restrict__ Cb,
                      int K, int lda,
                      const int* __restrict__ gidx) {
    __shared__ unsigned short As[2][128 * 32];
    __shared__ unsigned short Bs[2][128 * 32];

    const int tid  = threadIdx.x;
    const int m0   = blockIdx.y * 128;
    const int n0   = blockIdx.x * 128;
    const int lane = tid & 31;
    const int wave = tid >> 5;
    const int wm = wave >> 1, wn = wave & 1;
    const int l15 = lane & 15, lhalf = lane >> 4;

    v8f acc[2][4];
    #pragma unroll
    for (int i = 0; i < 2; i++)
        #pragma unroll
        for (int j = 0; j < 4; j++)
            #pragma unroll
            for (int e = 0; e < 8; e++) acc[i][j][e] = 0.0f;

    // staging slice: thread -> (row rowL, 32B half) for BOTH A and B tiles
    const int rowL = tid >> 1;
    const int half = tid & 1;
    long long arow;
    if (GATHER) {
        int r = m0 + rowL;
        int b = r >> 9, i = r & 511;
        arow = (long long)b * 1024 + gidx[b * 512 + i];
    } else {
        arow = m0 + rowL;
    }
    const unsigned short* aptr = A + arow * (long long)lda + half * 16;
    const unsigned short* bptr = Bt + (long long)(n0 + rowL) * K + half * 16;
    const int ldsOff = rowL * 32 + half * 16;

    const int nt = K >> 5;   // K/32 tiles

    // prologue: stage tile 0 into buffer 0
    {
        const uint4* ap = (const uint4*)aptr;
        const uint4* bp = (const uint4*)bptr;
        uint4 a0 = ap[0], a1 = ap[1];
        uint4 b0 = bp[0], b1 = bp[1];
        uint4* ad = (uint4*)(As[0] + ldsOff);
        uint4* bd = (uint4*)(Bs[0] + ldsOff);
        ad[0] = a0; ad[1] = a1;
        bd[0] = b0; bd[1] = b1;
    }

    for (int t = 0; t < nt; t++) {
        __syncthreads();   // buffer t&1 ready; buffer (t+1)&1 free

        // issue next tile's global loads early (latency hidden behind WMMAs)
        uint4 na0, na1, nb0, nb1;
        const bool more = (t + 1 < nt);
        if (more) {
            const uint4* ap = (const uint4*)(aptr + (t + 1) * 32);
            const uint4* bp = (const uint4*)(bptr + (t + 1) * 32);
            na0 = ap[0]; na1 = ap[1];
            nb0 = bp[0]; nb1 = bp[1];
            if (t + 2 < nt) {   // CDNA5 global_prefetch_b8 for t+2
                __builtin_prefetch((const void*)(aptr + (t + 2) * 32), 0, 1);
                __builtin_prefetch((const void*)(bptr + (t + 2) * 32), 0, 1);
            }
        }

        const unsigned short* Ab = As[t & 1];
        const unsigned short* Bb = Bs[t & 1];

        union U { uint4 q[2]; v16bf v; };
        // A frag (16-bit 16x32): lane<16 -> K {0..7,16..23}; lane>=16 -> K {8..15,24..31}
        U af[2];
        #pragma unroll
        for (int ms = 0; ms < 2; ms++) {
            const unsigned short* p = Ab + (wm * 32 + ms * 16 + l15) * 32 + lhalf * 8;
            af[ms].q[0] = *(const uint4*)p;
            af[ms].q[1] = *(const uint4*)(p + 16);
        }
        // B frag (32x16): lane<16 -> K {0..7,16..23}; lane>=16 -> K {8..15,24..31} (LDS holds B^T rows)
        U bfr[4];
        #pragma unroll
        for (int ns = 0; ns < 4; ns++) {
            const unsigned short* p = Bb + (wn * 64 + ns * 16 + l15) * 32 + lhalf * 8;
            bfr[ns].q[0] = *(const uint4*)p;
            bfr[ns].q[1] = *(const uint4*)(p + 16);
        }
        #pragma unroll
        for (int ns = 0; ns < 4; ns++)
            #pragma unroll
            for (int ms = 0; ms < 2; ms++)
                acc[ms][ns] = __builtin_amdgcn_wmma_f32_16x16x32_bf16(
                    false, af[ms].v, false, bfr[ns].v, (short)0, acc[ms][ns], false, false);

        // commit next tile into the other buffer
        if (more) {
            uint4* ad = (uint4*)(As[(t + 1) & 1] + ldsOff);
            uint4* bd = (uint4*)(Bs[(t + 1) & 1] + ldsOff);
            ad[0] = na0; ad[1] = na1;
            bd[0] = nb0; bd[1] = nb1;
        }
    }

    // epilogue: C/D layout -> lane: n = l15, m = lhalf*8 + vgpr
    #pragma unroll
    for (int ms = 0; ms < 2; ms++)
        #pragma unroll
        for (int ns = 0; ns < 4; ns++) {
            int nl = n0 + wn * 64 + ns * 16 + l15;
            float bv = bias[nl];
            #pragma unroll
            for (int g = 0; g < 8; g++) {
                long long r = m0 + wm * 32 + ms * 16 + lhalf * 8 + g;
                float v = acc[ms][ns][g] + bv;
                if (EPI == 0) {
                    Cf[r * 1024 + nl] = v;
                } else {
                    v = v >= 0.0f ? v : 0.2f * v;   // leaky(0.2)
                    Cb[r * 1024 + nl] = f2bf(v);
                }
            }
        }
}

// ---------------------------------------------------------------------------
// BatchNorm column stats: sum / sumsq over M rows of y[M,1024]
// ---------------------------------------------------------------------------
__global__ __launch_bounds__(256)
void bn_stats_kernel(const float* __restrict__ y, float* __restrict__ sum,
                     float* __restrict__ sq, int rowsPerChunk) {
    int c = blockIdx.x * 256 + threadIdx.x;
    long long r0 = (long long)blockIdx.y * rowsPerChunk;
    float s = 0.0f, q = 0.0f;
    for (int i = 0; i < rowsPerChunk; i++) {
        float v = y[(r0 + i) * 1024 + c];
        s += v; q += v * v;
    }
    atomicAdd(&sum[c], s);
    atomicAdd(&sq[c],  q);
}

__global__ void bn_finalize_kernel(const float* sum, const float* sq,
                                   const float* g, const float* b,
                                   float* scale, float* shift, float invM) {
    int c = blockIdx.x * 256 + threadIdx.x;
    float m   = sum[c] * invM;
    float var = sq[c] * invM - m * m;
    float sc  = g[c] * rsqrtf(var + 1e-5f);
    scale[c] = sc;
    shift[c] = b[c] - m * sc;
}

// normalize1: x = relu(bn1(y)) + 1e-8 -> bf16, plus pooled_clu = sum_n x
__global__ __launch_bounds__(256)
void normalize1_kernel(const float* __restrict__ y, const float* __restrict__ scale,
                       const float* __restrict__ shift, unsigned short* __restrict__ x1b,
                       float* __restrict__ pooledClu) {
    int c = blockIdx.x * 256 + threadIdx.x;
    int b = blockIdx.y;
    float s = scale[c], sh = shift[c], accu = 0.0f;
    for (int n = 0; n < Nn; n++) {
        long long o = ((long long)b * Nn + n) * 1024 + c;
        float v = fmaxf(y[o] * s + sh, 0.0f) + 1e-8f;
        x1b[o] = f2bf(v);
        accu += v;
    }
    pooledClu[b * 1024 + c] = accu;
}

// normalize2: s = relu(bn2(s_raw)); pooled_sel = sum_i s (s itself not stored)
__global__ __launch_bounds__(256)
void normalize2_kernel(const float* __restrict__ sraw, const float* __restrict__ scale,
                       const float* __restrict__ shift, float* __restrict__ pooledSel) {
    int c = blockIdx.x * 256 + threadIdx.x;
    int b = blockIdx.y;
    float s = scale[c], sh = shift[c], accu = 0.0f;
    for (int i = 0; i < KSEL; i++) {
        long long o = ((long long)b * KSEL + i) * 1024 + c;
        accu += fmaxf(sraw[o] * s + sh, 0.0f);
    }
    pooledSel[b * 1024 + c] = accu;
}

// attention = hh(bf16)[32768,1024] @ Wa2[1024] + ba2 ; one wave per row
__global__ __launch_bounds__(256)
void att_gemv_kernel(const unsigned short* __restrict__ hh, const float* __restrict__ wa2,
                     const float* __restrict__ ba2, float* __restrict__ att) {
    int lane = threadIdx.x & 31, wave = threadIdx.x >> 5;
    long long r = (long long)blockIdx.x * 8 + wave;
    float acc = 0.0f;
    #pragma unroll 4
    for (int i = 0; i < 32; i++) {
        int h = lane + i * 32;
        acc += bf2f(hh[r * 1024 + h]) * wa2[h];
    }
    for (int m = 16; m >= 1; m >>= 1) acc += __shfl_xor(acc, m, 32);
    if (lane == 0) att[r] = acc + ba2[0];
}

// stable top-k (k=512 of 1024) via rank counting; matches jax.lax.top_k order
__global__ __launch_bounds__(1024)
void topk_kernel(const float* __restrict__ att, int* __restrict__ idx) {
    __shared__ float sa[1024];
    int b = blockIdx.x, t = threadIdx.x;
    sa[t] = att[b * 1024 + t];
    __syncthreads();
    float ai = sa[t];
    int rank = 0;
    for (int j = 0; j < 1024; j++) {
        float aj = sa[j];
        rank += (aj > ai) || (aj == ai && j < t);
    }
    if (rank < KSEL) idx[b * KSEL + rank] = t;
}

// new_adj[b,i,j] = adj[b, idx[b,i], idx[b,j]]
__global__ __launch_bounds__(256)
void adj_gather_kernel(const float* __restrict__ adj, const int* __restrict__ idx,
                       float* __restrict__ outAdj) {
    int lin = blockIdx.x * 256 + threadIdx.x;   // 0 .. 512*512-1
    int b = blockIdx.y;
    int i = lin >> 9, j = lin & 511;
    int ri = idx[b * KSEL + i], rj = idx[b * KSEL + j];
    outAdj[((long long)b * KSEL + i) * KSEL + j] =
        adj[(long long)b * Nn * Nn + (long long)ri * Nn + rj];
}

// combined = concat(pooled_sel/512, pooled_clu/1024) + 1e-8
__global__ void combine_kernel(const float* __restrict__ ps, const float* __restrict__ pc,
                               float* __restrict__ comb) {
    int t = blockIdx.x * 256 + threadIdx.x;     // 32*2048
    int b = t >> 11, j = t & 2047;
    float v = (j < 1024) ? ps[b * 1024 + j] * (1.0f / 512.0f)
                         : pc[b * 1024 + (j - 1024)] * (1.0f / 1024.0f);
    comb[t] = v + 1e-8f;
}

__global__ void head_init_kernel(const float* bp2, const float* bf2v, float* out) {
    int t = threadIdx.x;
    if (t < 128) out[t] = bp2[t & 3];           // citation bias [32,4]
    if (t < 320) out[128 + t] = bf2v[t % 10];   // field bias    [32,10]
}

// head: h = relu(bn(comb @ W1 + b1)); out += h @ W2. One wave per column j.
template<int OUTN>
__global__ __launch_bounds__(32)
void head_kernel(const float* __restrict__ comb, const float* __restrict__ W1,
                 const float* __restrict__ b1, const float* __restrict__ g,
                 const float* __restrict__ bb, const float* __restrict__ W2,
                 float* __restrict__ out) {
    int j = blockIdx.x;       // 0..1023
    int r = threadIdx.x;      // 0..31 (batch row)
    float dot = b1[j];
    #pragma unroll 8
    for (int k = 0; k < 2048; k++) dot += comb[r * 2048 + k] * W1[k * 1024 + j];
    float s = dot, q = dot * dot;
    for (int m = 16; m >= 1; m >>= 1) { s += __shfl_xor(s, m, 32); q += __shfl_xor(q, m, 32); }
    float mean = s * (1.0f / 32.0f);
    float var  = q * (1.0f / 32.0f) - mean * mean;
    float h = g[j] * (dot - mean) * rsqrtf(var + 1e-5f) + bb[j];
    h = fmaxf(h, 0.0f);
    #pragma unroll
    for (int o = 0; o < OUTN; o++)
        atomicAdd(&out[r * OUTN + o], h * W2[j * OUTN + o]);
}

// ---------------------------------------------------------------------------
// Launcher
// ---------------------------------------------------------------------------
extern "C" void kernel_launch(void* const* d_in, const int* in_sizes, int n_in,
                              void* d_out, int out_size, void* d_ws, size_t ws_size,
                              hipStream_t stream) {
    const float* x     = (const float*)d_in[0];
    const float* adj   = (const float*)d_in[1];
    const float* We    = (const float*)d_in[2];
    const float* be    = (const float*)d_in[3];
    const float* bn1_g = (const float*)d_in[4];
    const float* bn1_b = (const float*)d_in[5];
    const float* Wa1   = (const float*)d_in[6];
    const float* ba1   = (const float*)d_in[7];
    const float* Wa2   = (const float*)d_in[8];
    const float* ba2   = (const float*)d_in[9];
    const float* Wc    = (const float*)d_in[10];
    const float* bc    = (const float*)d_in[11];
    const float* bn2_g = (const float*)d_in[12];
    const float* bn2_b = (const float*)d_in[13];
    const float* Wp1   = (const float*)d_in[14];
    const float* bp1   = (const float*)d_in[15];
    const float* bnp_g = (const float*)d_in[16];
    const float* bnp_b = (const float*)d_in[17];
    const float* Wp2   = (const float*)d_in[18];
    const float* bp2   = (const float*)d_in[19];
    const float* Wf1   = (const float*)d_in[20];
    const float* bf1   = (const float*)d_in[21];
    const float* bnf_g = (const float*)d_in[22];
    const float* bnf_b = (const float*)d_in[23];
    const float* Wf2   = (const float*)d_in[24];
    const float* bf2v  = (const float*)d_in[25];

    float* outp   = (float*)d_out;        // [0,128)=citation, [128,448)=field
    float* adjOut = outp + 448;           // [448, 448+32*512*512)

    // ---- workspace carve-out (256B aligned) ----
    char* w = (char*)d_ws;
    size_t off = 0;
    auto take = [&](size_t bytes) -> void* {
        void* p = w + off;
        off = (off + bytes + 255) & ~(size_t)255;
        return p;
    };
    unsigned short* xb   = (unsigned short*)take((size_t)Bsz * Nn * DIN * 2); // 50.3 MB
    float*          y    = (float*)take((size_t)Bsz * Nn * Hh * 4);           // 134 MB
    unsigned short* hh   = (unsigned short*)y;                                // alias (y dead)
    float*          sraw = y + (size_t)Bsz * KSEL * Hh;                       // alias upper half
    unsigned short* x1b  = (unsigned short*)take((size_t)Bsz * Nn * Hh * 2);  // 67 MB
    unsigned short* WeT  = (unsigned short*)take((size_t)DIN * Hh * 2);       // [1024,768]
    unsigned short* Wa1T = (unsigned short*)take((size_t)Hh * Hh * 2);        // [1024,1024]
    unsigned short* WcT  = (unsigned short*)take((size_t)Hh * Hh * 2);        // [1024,1024]
    float* att   = (float*)take((size_t)Bsz * Nn * 4);
    int*   idxb  = (int*)take((size_t)Bsz * KSEL * 4);
    float* bnbuf = (float*)take(8192 * 4);
    float* bn1sum = bnbuf,        *bn1sq = bnbuf + 1024;
    float* sc1 = bnbuf + 2048,    *sh1   = bnbuf + 3072;
    float* bn2sum = bnbuf + 4096, *bn2sq = bnbuf + 5120;
    float* sc2 = bnbuf + 6144,    *sh2   = bnbuf + 7168;
    float* pooledSel = (float*)take((size_t)Bsz * Hh * 4);
    float* pooledClu = (float*)take((size_t)Bsz * Hh * 4);
    float* comb      = (float*)take((size_t)Bsz * 2 * Hh * 4);

    // ---- 0. zero accumulators (deterministic across replays) ----
    zero_kernel<<<32, 256, 0, stream>>>(bnbuf, 8192);

    // ---- 1. bf16 conversions: activations row-major, weights transposed ----
    cvt_bf16_kernel<<<2048, 256, 0, stream>>>(x, xb, (long long)Bsz * Nn * DIN);
    transpose_cvt_kernel<<<dim3(32, 24), 256, 0, stream>>>(We,  WeT,  DIN, Hh);
    transpose_cvt_kernel<<<dim3(32, 32), 256, 0, stream>>>(Wa1, Wa1T, Hh,  Hh);
    transpose_cvt_kernel<<<dim3(32, 32), 256, 0, stream>>>(Wc,  WcT,  Hh,  Hh);

    // ---- 2. embed GEMM: y = x @ We + be ----
    gemm_bf16_kernel<0, false><<<dim3(8, 256), 256, 0, stream>>>(
        xb, WeT, be, y, nullptr, DIN, DIN, nullptr);

    // ---- 3. bn1 + relu + 1e-8 -> x1b (bf16), pooled_clu ----
    bn_stats_kernel<<<dim3(4, 32), 256, 0, stream>>>(y, bn1sum, bn1sq, 1024);
    bn_finalize_kernel<<<4, 256, 0, stream>>>(bn1sum, bn1sq, bn1_g, bn1_b, sc1, sh1,
                                              1.0f / (Bsz * Nn));
    normalize1_kernel<<<dim3(4, 32), 256, 0, stream>>>(y, sc1, sh1, x1b, pooledClu);

    // ---- 4. attention: hh = leaky(x1 @ Wa1 + ba1) (bf16), att = hh @ Wa2 + ba2 ----
    gemm_bf16_kernel<1, false><<<dim3(8, 256), 256, 0, stream>>>(
        x1b, Wa1T, ba1, nullptr, hh, Hh, Hh, nullptr);
    att_gemv_kernel<<<4096, 256, 0, stream>>>(hh, Wa2, ba2, att);

    // ---- 5. top-k + adjacency reindex (straight to d_out) ----
    topk_kernel<<<32, 1024, 0, stream>>>(att, idxb);
    adj_gather_kernel<<<dim3(1024, 32), 256, 0, stream>>>(adj, idxb, adjOut);

    // ---- 6. conv GEMM on gathered rows: s_raw = sel_x @ Wc + bc ----
    gemm_bf16_kernel<0, true><<<dim3(8, 128), 256, 0, stream>>>(
        x1b, WcT, bc, sraw, nullptr, Hh, Hh, idxb);

    // ---- 7. bn2 + relu -> pooled_sel ----
    bn_stats_kernel<<<dim3(4, 32), 256, 0, stream>>>(sraw, bn2sum, bn2sq, 512);
    bn_finalize_kernel<<<4, 256, 0, stream>>>(bn2sum, bn2sq, bn2_g, bn2_b, sc2, sh2,
                                              1.0f / (Bsz * KSEL));
    normalize2_kernel<<<dim3(4, 32), 256, 0, stream>>>(sraw, sc2, sh2, pooledSel);

    // ---- 8. combined + heads ----
    combine_kernel<<<256, 256, 0, stream>>>(pooledSel, pooledClu, comb);
    head_init_kernel<<<1, 512, 0, stream>>>(bp2, bf2v, outp);
    head_kernel<4><<<1024, 32, 0, stream>>>(comb, Wp1, bp1, bnp_g, bnp_b, Wp2, outp);
    head_kernel<10><<<1024, 32, 0, stream>>>(comb, Wf1, bf1, bnf_g, bnf_b, Wf2, outp + 128);
}
